// SwinTransformerEncoder_18021682774782
// MI455X (gfx1250) — matrix operand order repaired and checked by reading, not compiled, not run on hardware
//
#include <hip/hip_runtime.h>
#include <hip/hip_bf16.h>
#include <stdint.h>

#if defined(__has_builtin)
#if __has_builtin(__builtin_amdgcn_tensor_load_to_lds)
#define HAVE_TDM 1
#endif
#endif
#ifndef HAVE_TDM
#define HAVE_TDM 0
#endif

// ---------------- types & helpers ----------------
typedef __bf16 bf16;
typedef __attribute__((ext_vector_type(16))) __bf16 v16bf;
typedef __attribute__((ext_vector_type(8)))  __bf16 v8bf;
typedef __attribute__((ext_vector_type(8)))  float  v8f;
typedef __attribute__((ext_vector_type(4)))  uint32_t v4u;
typedef __attribute__((ext_vector_type(8)))  int      v8i;
typedef __attribute__((ext_vector_type(4)))  int      v4i;

#define WMMA_BF16(a, b, c) \
  __builtin_amdgcn_wmma_f32_16x16x32_bf16(false, (a), false, (b), (short)0, (c), false, false)

__device__ __forceinline__ bf16 f2bf(float f) {
  union { float f; uint32_t u; } x; x.f = f;
  uint32_t r = (x.u + 0x7FFFu + ((x.u >> 16) & 1u)) >> 16;   // RNE
  union { uint16_t u; bf16 b; } o; o.u = (uint16_t)r;
  return o.b;
}
__device__ __forceinline__ v16bf mk16(v8bf lo, v8bf hi) {
  v16bf r;
#pragma unroll
  for (int i = 0; i < 8; ++i) { r[i] = lo[i]; r[i + 8] = hi[i]; }
  return r;
}

#if HAVE_TDM
// Issue a TDM 2D tile load (rows x cols bf16, row stride strideElems) into LDS.
// D# layout per CDNA5 ISA ch.8: group0 = {count/type/addr}, group1 = dims/strides.
// This toolchain exposes the 6-arg builtin: (v4u, v8i, v4i, v4i, v8i, i32 cpol).
__device__ __forceinline__ void tdm_load_2d(const bf16* g, int strideElems, int rows,
                                            int cols, void* lds) {
  uint64_t ga = (uint64_t)(uintptr_t)g;
  uint32_t la = (uint32_t)(uintptr_t)lds;          // low 32 bits = LDS byte offset
  v4u g0;
  g0[0] = 1u;                                       // count=1 (valid), user mode
  g0[1] = la;                                       // lds_addr
  g0[2] = (uint32_t)ga;                             // global_addr[31:0]
  g0[3] = (uint32_t)((ga >> 32) & 0x01FFFFFFu) | 0x80000000u;  // addr[56:32] | type=2
  v8i g1;
  g1[0] = 0x00010000;                               // data_size = 1 -> 2 bytes
  g1[1] = (int)(((uint32_t)strideElems & 0xFFFFu) << 16);            // tensor_dim0 lo16
  g1[2] = (int)((((uint32_t)strideElems >> 16) & 0xFFFFu) |
                (((uint32_t)rows & 0xFFFFu) << 16));                 // dim0 hi | dim1 lo
  g1[3] = (int)(((uint32_t)cols & 0xFFFFu) << 16);                   // dim1 hi=0 | tile_dim0
  g1[4] = (int)((uint32_t)rows & 0xFFFFu);                           // tile_dim1 | tile_dim2=0
  g1[5] = strideElems;                                               // dim0_stride[31:0]
  g1[6] = 0;                                                         // stride hi / dim1_stride
  g1[7] = 0;
  v4i z4 = {0, 0, 0, 0};
  v8i z8 = {0, 0, 0, 0, 0, 0, 0, 0};
  __builtin_amdgcn_tensor_load_to_lds(g0, g1, z4, z4, z8, 0);
}
#endif

// ---------------- problem constants ----------------
constexpr int cB = 16, cH = 56, cW = 56, cC = 384;
constexpr int cNH = 12, cWS = 7, cN = 49, cHD = 32;
constexpr int NWIN = cB * 64;          // 1024 windows
constexpr int TOK  = cB * cH * cW;     // 50176 tokens (divisible by 64)
constexpr float SCALE = 0.17677669529663687f; // 32^-0.5

// ---------------- misc kernels ----------------
__global__ void cvt_f32_bf16(const float* __restrict__ in, bf16* __restrict__ out, int n) {
  int i = blockIdx.x * 256 + threadIdx.x;
  if (i < n) out[i] = f2bf(in[i]);
}

__global__ void fill0_u32(uint32_t* __restrict__ p, int n) {
  int i = blockIdx.x * 256 + threadIdx.x;
  if (i < n) p[i] = 0u;
}

// Fused attention bias: biasAll[nwin][head][row64][col64] = rpb + mask, -1e30 padding.
__global__ void build_bias_kernel(const float* __restrict__ rpb, const int* __restrict__ relidx,
                                  const float* __restrict__ mask, float* __restrict__ out,
                                  int use_mask) {
  int idx = blockIdx.x * 256 + threadIdx.x;      // 64*12*64*64 = 3,145,728
  int col = idx & 63;
  int row = (idx >> 6) & 63;
  int h   = (idx >> 12) % cNH;
  int nw  = idx / (cNH << 12);
  float vv = -1e30f;
  if (row < cN && col < cN) {
    vv = rpb[relidx[row * cN + col] * cNH + h];
    if (use_mask) vv += mask[((size_t)nw * cN + row) * cN + col];
  }
  out[idx] = vv;
}

// LayerNorm over C=384, one wave per token; optional shift+window-partition.
__launch_bounds__(256)
__global__ void ln_kernel(const float* __restrict__ x, const float* __restrict__ g,
                          const float* __restrict__ bta, bf16* __restrict__ out,
                          int mode_shift) {
  const int wave = threadIdx.x >> 5, lane = threadIdx.x & 31;
  const int t = blockIdx.x * 8 + wave;
  const float* src;
  if (mode_shift >= 0) {
    int win = t / cN, n = t - win * cN;
    int b = win >> 6, wi = win & 63;
    int hr = (wi >> 3) * cWS + n / cWS;
    int wr = (wi & 7) * cWS + n % cWS;
    int hs = hr + mode_shift; if (hs >= cH) hs -= cH;
    int wsc = wr + mode_shift; if (wsc >= cW) wsc -= cW;
    src = x + ((size_t)(b * cH + hs) * cW + wsc) * cC;
  } else {
    src = x + (size_t)t * cC;
  }
  float vals[12];
  float s = 0.f, sq = 0.f;
#pragma unroll
  for (int j = 0; j < 12; ++j) {
    float v = src[j * 32 + lane];
    vals[j] = v; s += v; sq += v * v;
  }
#pragma unroll
  for (int off = 16; off; off >>= 1) {
    s  += __shfl_xor(s, off, 32);
    sq += __shfl_xor(sq, off, 32);
  }
  const float mean = s * (1.f / cC);
  const float var  = sq * (1.f / cC) - mean * mean;
  const float inv  = rsqrtf(var + 1e-5f);
  bf16* dst = out + (size_t)t * cC;
#pragma unroll
  for (int j = 0; j < 12; ++j) {
    int c = j * 32 + lane;
    dst[c] = f2bf((vals[j] - mean) * inv * g[c] + bta[c]);
  }
}

// reverse window partition + roll(+shift) + residual add, elementwise.
__launch_bounds__(256)
__global__ void rev_add_kernel(const float* __restrict__ tmp, float* __restrict__ x, int shift) {
  int idx = blockIdx.x * 256 + threadIdx.x;
  int t = idx / cC, c = idx - t * cC;
  int b = t / (cH * cW), rem = t - b * (cH * cW);
  int h_ = rem / cW, w_ = rem - h_ * cW;
  int hr = h_ - shift; if (hr < 0) hr += cH;
  int wr = w_ - shift; if (wr < 0) wr += cW;
  int win = (b << 6) + (hr / cWS) * 8 + (wr / cWS);
  int n = (hr % cWS) * cWS + (wr % cWS);
  x[idx] += tmp[((size_t)win * cN + n) * cC + c];
}

// ---------------- generic bf16 WMMA GEMM: Y = A[M,K] * W[N,K]^T ----------------
enum { EPI_QKV = 0, EPI_GELU = 1, EPI_F32 = 2, EPI_RES = 3 };

template <int EPI>
__launch_bounds__(256)
__global__ void gemm_kernel(const bf16* __restrict__ A, const bf16* __restrict__ Wt,
                            const float* __restrict__ bias,
                            bf16* __restrict__ ybf, float* __restrict__ yf,
                            float* __restrict__ resid,
                            bf16* __restrict__ qb, bf16* __restrict__ kb, bf16* __restrict__ vb,
                            int N, int K, float qscale) {
#if HAVE_TDM
  __shared__ __align__(128) bf16 As[2][64][32];
  __shared__ __align__(128) bf16 Bs[2][128][32];
#else
  __shared__ __align__(128) bf16 As[1][64][32];
  __shared__ __align__(128) bf16 Bs[1][128][32];
#endif
  const int tid = threadIdx.x;
  const int wave = tid >> 5, lane = tid & 31;
  const int lh = lane & 15, hi = lane >> 4;
  const int wm = (wave >> 2) * 32, wn = (wave & 3) * 32;
  const int bm = blockIdx.y, bn = blockIdx.x;
  const int koff = hi ? 8 : 0, bko = hi ? 16 : 0;
  v8f acc[2][2] = {};

  auto compute = [&](int buf) {
    v16bf af[2], bfg[2];
#pragma unroll
    for (int i = 0; i < 2; ++i) {
      af[i] = *(const v16bf*)(&As[buf][wm + i * 16 + lh][bko]);
    }
#pragma unroll
    for (int j = 0; j < 2; ++j)
      bfg[j] = *(const v16bf*)(&Bs[buf][wn + j * 16 + lh][bko]);
#pragma unroll
    for (int i = 0; i < 2; ++i)
#pragma unroll
      for (int j = 0; j < 2; ++j)
        acc[i][j] = WMMA_BF16(af[i], bfg[j], acc[i][j]);
  };

  const int S = K >> 5;
#if HAVE_TDM
  // Double-buffered Tensor-Data-Mover pipeline: wave 0 issues async tile loads,
  // all waves compute WMMAs on the other buffer. s_wait_tensorcnt + barrier order it.
  const bf16* abase = A  + (size_t)(bm * 64)  * K;
  const bf16* bbase = Wt + (size_t)(bn * 128) * K;
  if (wave == 0) {
    tdm_load_2d(abase, K, 64, 32, &As[0][0][0]);
    tdm_load_2d(bbase, K, 128, 32, &Bs[0][0][0]);
  }
  int buf = 0;
  for (int s = 0; s < S; ++s) {
    __builtin_amdgcn_s_wait_tensorcnt(0);
    __syncthreads();
    if (wave == 0 && s + 1 < S) {
      tdm_load_2d(abase + (s + 1) * 32, K, 64, 32, &As[buf ^ 1][0][0]);
      tdm_load_2d(bbase + (s + 1) * 32, K, 128, 32, &Bs[buf ^ 1][0][0]);
    }
    compute(buf);
    buf ^= 1;
  }
  __syncthreads();
#else
  const int ar = tid >> 2, ac = (tid & 3) * 8;
  const int br = tid >> 1, bc = (tid & 1) * 16;
  const bf16* aptr = A  + (size_t)(bm * 64  + ar) * K + ac;
  const bf16* bptr = Wt + (size_t)(bn * 128 + br) * K + bc;
  for (int s = 0; s < S; ++s) {
    const int k0 = s * 32;
    *(v8bf*)&As[0][ar][ac]     = *(const v8bf*)(aptr + k0);
    *(v8bf*)&Bs[0][br][bc]     = *(const v8bf*)(bptr + k0);
    *(v8bf*)&Bs[0][br][bc + 8] = *(const v8bf*)(bptr + k0 + 8);
    if (k0 + 32 < K) {
      __builtin_prefetch(aptr + k0 + 32, 0, 3);
      __builtin_prefetch(bptr + k0 + 32, 0, 3);
    }
    __syncthreads();
    compute(0);
    __syncthreads();
  }
#endif

#pragma unroll
  for (int i = 0; i < 2; ++i) {
#pragma unroll
    for (int j = 0; j < 2; ++j) {
      const int col = bn * 128 + wn + j * 16 + lh;
      const float bv = bias[col];
#pragma unroll
      for (int e = 0; e < 8; ++e) {
        const int row = bm * 64 + wm + i * 16 + hi * 8 + e;
        float val = acc[i][j][e] + bv;
        if (EPI == EPI_QKV) {
          const int s  = col / cC;
          const int hh = (col % cC) >> 5;
          const int d  = col & 31;
          if (s == 0) val *= qscale;
          bf16* dst = (s == 0) ? qb : (s == 1) ? kb : vb;
          const int w2 = row / cN, nn = row - w2 * cN;
          dst[(((size_t)w2 * cNH + hh) * 64 + nn) * cHD + d] = f2bf(val);
        } else if (EPI == EPI_GELU) {
          val = 0.5f * val * (1.f + erff(val * 0.70710678118654752f));
          ybf[(size_t)row * N + col] = f2bf(val);
        } else if (EPI == EPI_F32) {
          yf[(size_t)row * N + col] = val;
        } else {
          resid[(size_t)row * N + col] += val;
        }
      }
    }
  }
}

// ---------------- fused window attention: one block per (window, head) ----------------
__launch_bounds__(128)
__global__ void attn_kernel(const bf16* __restrict__ q, const bf16* __restrict__ k,
                            const bf16* __restrict__ v, const float* __restrict__ biasAll,
                            bf16* __restrict__ out) {
  __shared__ __align__(64) bf16 Ks[64][32];
  __shared__ __align__(64) bf16 VsT[32][64];
  __shared__ __align__(64) float Sm[64][65];
  __shared__ __align__(64) bf16 Ps[64][64];

  const int blk = blockIdx.x;
  const int win = blk / cNH, h = blk - win * cNH;
  const int tid = threadIdx.x, wv = tid >> 5, lane = tid & 31;
  const int lh = lane & 15, hiq = lane >> 4;
  const size_t base = ((size_t)win * cNH + h) * 64 * cHD;

#if HAVE_TDM
  if (wv == 0)   // K tile is one contiguous 4KB block: 1-row TDM transfer
    tdm_load_2d(k + base, 2048, 1, 2048, &Ks[0][0]);
#else
  {
    int r = tid >> 1, c = (tid & 1) * 16;
    *(v8bf*)&Ks[r][c]     = *(const v8bf*)(k + base + r * cHD + c);
    *(v8bf*)&Ks[r][c + 8] = *(const v8bf*)(k + base + r * cHD + c + 8);
  }
#endif
  { // V tile transposed -> LDS (so B-fragments of P@V are contiguous rows)
    int tok = tid >> 1, dg = (tid & 1) * 16;
    const bf16* src = v + base + tok * cHD + dg;
#pragma unroll
    for (int j = 0; j < 16; ++j) VsT[dg + j][tok] = src[j];
  }
  // q A-fragment straight from global (rows 16*wv..16*wv+15, K = HD = 32)
  const int koff = hiq ? 8 : 0;
  const bf16* qrow = q + base + (size_t)(16 * wv + lh) * cHD;
  const v16bf afrag = *(const v16bf*)(qrow + (hiq ? 16 : 0));
#if HAVE_TDM
  __builtin_amdgcn_s_wait_tensorcnt(0);
#endif
  __syncthreads();

  v8f sacc[4];
#pragma unroll
  for (int nt = 0; nt < 4; ++nt) {
    v16bf bfrag = *(const v16bf*)(&Ks[nt * 16 + lh][hiq ? 16 : 0]);
    v8f z = {};
    sacc[nt] = WMMA_BF16(afrag, bfrag, z);
  }

  // S + fused (rpb+mask+padding) bias -> LDS, branch-free
  const float* bptr = biasAll + (((size_t)(win & 63) * cNH + h) << 12);
#pragma unroll
  for (int nt = 0; nt < 4; ++nt) {
    const int col = nt * 16 + lh;
#pragma unroll
    for (int e = 0; e < 8; ++e) {
      const int row = 16 * wv + 8 * hiq + e;
      Sm[row][col] = sacc[nt][e] + bptr[(row << 6) + col];
    }
  }
  __syncthreads();

  if (tid < 64) { // row-wise softmax over 49 valid columns
    const int r = tid;
    if (r < cN) {
      float mx = -1e30f;
      for (int c = 0; c < cN; ++c) mx = fmaxf(mx, Sm[r][c]);
      float s = 0.f;
      for (int c = 0; c < cN; ++c) s += __expf(Sm[r][c] - mx);
      const float invs = 1.f / s;
      for (int c = 0; c < 64; ++c)
        Ps[r][c] = f2bf((c < cN) ? __expf(Sm[r][c] - mx) * invs : 0.f);
    } else {
      for (int c = 0; c < 64; ++c) Ps[r][c] = f2bf(0.f);
    }
  }
  __syncthreads();

  v8f oacc[2] = {};
#pragma unroll
  for (int ks = 0; ks < 2; ++ks) {
    v16bf pa = *(const v16bf*)(&Ps[16 * wv + lh][ks * 32 + (hiq ? 16 : 0)]);
#pragma unroll
    for (int nt = 0; nt < 2; ++nt) {
      v16bf bv = *(const v16bf*)(&VsT[nt * 16 + lh][ks * 32 + (hiq ? 16 : 0)]);
      oacc[nt] = WMMA_BF16(pa, bv, oacc[nt]);
    }
  }
#pragma unroll
  for (int nt = 0; nt < 2; ++nt) {
    const int d = nt * 16 + lh;
#pragma unroll
    for (int e = 0; e < 8; ++e) {
      const int row = 16 * wv + 8 * hiq + e;
      if (row < cN)
        out[(((size_t)win * cN + row) * cNH + h) * cHD + d] = f2bf(oacc[nt][e]);
    }
  }
}

// ---------------- launch ----------------
extern "C" void kernel_launch(void* const* d_in, const int* in_sizes, int n_in,
                              void* d_out, int out_size, void* d_ws, size_t ws_size,
                              hipStream_t stream) {
  const float* x_in   = (const float*)d_in[0];
  const float* mask   = (const float*)d_in[1];
  const int*   relidx = (const int*)d_in[2];
  const float* n1g = (const float*)d_in[3];
  const float* n1b = (const float*)d_in[4];
  const float* qkvw = (const float*)d_in[5];
  const float* qkvb = (const float*)d_in[6];
  const float* rpb  = (const float*)d_in[7];
  const float* projw = (const float*)d_in[8];
  const float* projb = (const float*)d_in[9];
  const float* n2g = (const float*)d_in[10];
  const float* n2b = (const float*)d_in[11];
  const float* fc1w = (const float*)d_in[12];
  const float* fc1b = (const float*)d_in[13];
  const float* fc2w = (const float*)d_in[14];
  const float* fc2b = (const float*)d_in[15];

  char* ws = (char*)d_ws;
  size_t off = 0;
  auto alloc = [&](size_t bytes) -> char* {
    char* p = ws + off; off += (bytes + 255) & ~(size_t)255; return p;
  };
  const size_t qkv_elems = (size_t)NWIN * cNH * 64 * cHD;     // padded per-head tiles
  float* x_ws   = (float*)alloc((size_t)TOK * cC * 4);
  bf16*  xw     = (bf16*) alloc((size_t)TOK * cC * 2);        // LN out / attn out / LN2 out
  bf16*  qb     = (bf16*) alloc(qkv_elems * 2);
  bf16*  kb     = (bf16*) alloc(qkv_elems * 2);
  bf16*  vb     = (bf16*) alloc(qkv_elems * 2);
  float* tmpf   = (float*)alloc((size_t)TOK * cC * 4);
  bf16*  hbuf   = (bf16*) alloc((size_t)TOK * 4 * cC * 2);
  float* biasA  = (float*)alloc((size_t)64 * cNH * 64 * 64 * 4);
  bf16*  qkvw_b = (bf16*) alloc((size_t)2 * 3 * cC * cC * 2);
  bf16*  projw_b= (bf16*) alloc((size_t)2 * cC * cC * 2);
  bf16*  fc1w_b = (bf16*) alloc((size_t)2 * 4 * cC * cC * 2);
  bf16*  fc2w_b = (bf16*) alloc((size_t)2 * 4 * cC * cC * 2);

  (void)hipMemcpyAsync(x_ws, x_in, (size_t)TOK * cC * 4, hipMemcpyDeviceToDevice, stream);

  auto cvt = [&](const float* in, bf16* o, int n) {
    cvt_f32_bf16<<<(n + 255) / 256, 256, 0, stream>>>(in, o, n);
  };
  cvt(qkvw,  qkvw_b,  2 * 3 * cC * cC);
  cvt(projw, projw_b, 2 * cC * cC);
  cvt(fc1w,  fc1w_b,  2 * 4 * cC * cC);
  cvt(fc2w,  fc2w_b,  2 * 4 * cC * cC);

  const int nz = (int)(qkv_elems / 2);  // u32 count per buffer
  const int nbias = 64 * cNH * 64 * 64;
  for (int i = 0; i < 2; ++i) {
    const int shift = (i == 0) ? 0 : (cWS / 2);
    fill0_u32<<<(nz + 255) / 256, 256, 0, stream>>>((uint32_t*)qb, nz);
    fill0_u32<<<(nz + 255) / 256, 256, 0, stream>>>((uint32_t*)kb, nz);
    fill0_u32<<<(nz + 255) / 256, 256, 0, stream>>>((uint32_t*)vb, nz);
    build_bias_kernel<<<nbias / 256, 256, 0, stream>>>(
        rpb + (size_t)i * 169 * cNH, relidx, mask, biasA, shift > 0 ? 1 : 0);

    ln_kernel<<<TOK / 8, 256, 0, stream>>>(x_ws, n1g + i * cC, n1b + i * cC, xw, shift);

    gemm_kernel<EPI_QKV><<<dim3(3 * cC / 128, TOK / 64), 256, 0, stream>>>(
        xw, qkvw_b + (size_t)i * 3 * cC * cC, qkvb + i * 3 * cC,
        nullptr, nullptr, nullptr, qb, kb, vb, 3 * cC, cC, SCALE);

    attn_kernel<<<NWIN * cNH, 128, 0, stream>>>(qb, kb, vb, biasA, xw);

    gemm_kernel<EPI_F32><<<dim3(cC / 128, TOK / 64), 256, 0, stream>>>(
        xw, projw_b + (size_t)i * cC * cC, projb + i * cC,
        nullptr, tmpf, nullptr, nullptr, nullptr, nullptr, cC, cC, 0.f);

    rev_add_kernel<<<(TOK * cC) / 256, 256, 0, stream>>>(tmpf, x_ws, shift);

    ln_kernel<<<TOK / 8, 256, 0, stream>>>(x_ws, n2g + i * cC, n2b + i * cC, xw, -1);

    gemm_kernel<EPI_GELU><<<dim3(4 * cC / 128, TOK / 64), 256, 0, stream>>>(
        xw, fc1w_b + (size_t)i * 4 * cC * cC, fc1b + i * 4 * cC,
        hbuf, nullptr, nullptr, nullptr, nullptr, nullptr, 4 * cC, cC, 0.f);

    gemm_kernel<EPI_RES><<<dim3(cC / 128, TOK / 64), 256, 0, stream>>>(
        hbuf, fc2w_b + (size_t)i * cC * 4 * cC, fc2b + i * cC,
        nullptr, nullptr, x_ws, nullptr, nullptr, nullptr, cC, 4 * cC, 0.f);
  }

  (void)hipMemcpyAsync(d_out, x_ws, (size_t)TOK * cC * 4, hipMemcpyDeviceToDevice, stream);
}
